// AnchorCrossAttnBlock_83674552861083
// MI455X (gfx1250) — matrix operand order repaired and checked by reading, not compiled, not run on hardware
//
#include <hip/hip_runtime.h>
#include <hip/hip_bf16.h>

typedef __bf16 bf16_t;
typedef __attribute__((ext_vector_type(16))) __bf16 v16bf;
typedef __attribute__((ext_vector_type(8)))  __bf16 bf16x8;
typedef __attribute__((ext_vector_type(4)))  __bf16 bf16x4;
typedef __attribute__((ext_vector_type(8)))  float  v8f;
typedef __attribute__((ext_vector_type(4)))  float  f32x4;

#define THREADS 256
#define MB 128
#define NB 64
#define KB 64     // K-tile: two WMMA k-steps per barrier pair
#define APAD 72   // padded LDS row (halves): 144B stride, 16B aligned, conflict-free
#define BPAD 72

#define BATCH 16
#define CC1   256
#define HW    4096
#define NANCH 256
#define KIM   8192
#define RATIO 0.0625f   // 256^-0.5

// ---------------------------------------------------------------------------
// Group-norm statistics -> per-(b,c) scale/shift  (y = x*scale + shift)
// blockIdx: x = group (0..31), y = batch, z = which norm {0:x,1:p,2:concat}
// ---------------------------------------------------------------------------
__global__ void gn_stats_kernel(const float* __restrict__ x, const float* __restrict__ p,
                                const float* nq_g, const float* nq_b,
                                const float* nkv_g, const float* nkv_b,
                                const float* na_g, const float* na_b,
                                float* sx, float* tx, float* sp, float* tp,
                                float* st, float* tt) {
  const int g = blockIdx.x, b = blockIdx.y, which = blockIdx.z;
  const int cpg = (which == 2) ? 16 : 8;
  const int C   = (which == 2) ? 512 : 256;
  const float* gam = (which == 0) ? nq_g : (which == 1) ? nkv_g : na_g;
  const float* bet = (which == 0) ? nq_b : (which == 1) ? nkv_b : na_b;
  float* Sout = (which == 0) ? sx : (which == 1) ? sp : st;
  float* Tout = (which == 0) ? tx : (which == 1) ? tp : tt;
  const int c0 = g * cpg;
  const int count = cpg * HW;

  float sum = 0.f, sq = 0.f;
  for (int i = threadIdx.x * 4; i < count; i += THREADS * 4) {
    int cc = c0 + (i >> 12);
    int n  = i & 4095;
    const float* src; int ch;
    if (which == 0)      { src = x; ch = cc; }
    else if (which == 1) { src = p; ch = cc; }
    else { if (cc < 256) { src = x; ch = cc; } else { src = p; ch = cc - 256; } }
    f32x4 v = *(const f32x4*)(src + ((size_t)b * CC1 + ch) * HW + n);
#pragma unroll
    for (int e = 0; e < 4; ++e) { sum += v[e]; sq += v[e] * v[e]; }
  }
  __shared__ float s1[THREADS], s2[THREADS];
  s1[threadIdx.x] = sum; s2[threadIdx.x] = sq;
  __syncthreads();
  for (int off = THREADS / 2; off > 0; off >>= 1) {
    if ((int)threadIdx.x < off) {
      s1[threadIdx.x] += s1[threadIdx.x + off];
      s2[threadIdx.x] += s2[threadIdx.x + off];
    }
    __syncthreads();
  }
  if ((int)threadIdx.x < cpg) {
    float mean = s1[0] / (float)count;
    float var  = s2[0] / (float)count - mean * mean;
    float rs   = rsqrtf(var + 1e-6f);
    int c = c0 + threadIdx.x;
    float sc = gam[c] * rs;
    Sout[b * C + c] = sc;
    Tout[b * C + c] = bet[c] - mean * sc;
  }
}

// ---------------------------------------------------------------------------
// WMMA tile core. As: [128][APAD] (row-major M x K). Bs: [64][BPAD] (col n
// holds its K contiguously). 8 waves; wave (wm,wn) owns 32x32 = 2x2 tiles.
// Two k-halves (ks = 0, 32) per call -> 8 WMMAs per wave per barrier pair.
// Fragment layouts per CDNA5 ISA 7.12.2 (wave32).
// ---------------------------------------------------------------------------
__device__ __forceinline__ void wmma_block_compute(const bf16_t* As, const bf16_t* Bs,
                                                   v8f acc[2][2]) {
  const int lane = threadIdx.x & 31;
  const int wid  = threadIdx.x >> 5;
  const int wm = wid >> 1, wn = wid & 1;
  const int row = lane & 15;
  const int hi  = lane >> 4;

#pragma unroll
  for (int ks = 0; ks < KB; ks += 32) {
    v16bf afr[2], bfr[2];
#pragma unroll
    for (int im = 0; im < 2; ++im) {
      const bf16_t* pa = As + (size_t)(wm * 32 + im * 16 + row) * APAD + ks;
      bf16x8 lo = *(const bf16x8*)(pa + hi * 8);        // K = ks+hi*8 .. +7
      bf16x8 hh = *(const bf16x8*)(pa + 16 + hi * 8);   // K = ks+16+hi*8 .. +7
      afr[im] = __builtin_shufflevector(lo, hh, 0, 1, 2, 3, 4, 5, 6, 7,
                                                8, 9, 10, 11, 12, 13, 14, 15);
    }
#pragma unroll
    for (int in = 0; in < 2; ++in) {
      const bf16_t* pb = Bs + (size_t)(wn * 32 + in * 16 + row) * BPAD + ks;
      bf16x8 lo = *(const bf16x8*)(pb + hi * 8);
      bf16x8 hh = *(const bf16x8*)(pb + 16 + hi * 8);
      bfr[in] = __builtin_shufflevector(lo, hh, 0, 1, 2, 3, 4, 5, 6, 7,
                                                8, 9, 10, 11, 12, 13, 14, 15);
    }
#pragma unroll
    for (int im = 0; im < 2; ++im)
#pragma unroll
      for (int in = 0; in < 2; ++in)
        acc[im][in] = __builtin_amdgcn_wmma_f32_16x16x32_bf16(
            false, afr[im], false, bfr[in], (short)0, acc[im][in], false, false);
  }
}

__device__ __forceinline__ void zero_acc(v8f acc[2][2]) {
#pragma unroll
  for (int im = 0; im < 2; ++im)
#pragma unroll
    for (int in = 0; in < 2; ++in)
#pragma unroll
      for (int e = 0; e < 8; ++e) acc[im][in][e] = 0.0f;
}

// --------------------------- tile loaders ----------------------------------
// A tile: f32 [M][K] -> LDS [128][APAD] bf16 (128x64 elements)
__device__ __forceinline__ void fill_A_f32(bf16_t* As, const float* A, int m0, int k0, int lda) {
#pragma unroll
  for (int j = 0; j < 8; ++j) {
    int idx = threadIdx.x + j * THREADS;   // 2048 float4 chunks
    int r = idx >> 4, c4 = idx & 15;
    f32x4 v = *(const f32x4*)(A + (size_t)(m0 + r) * lda + k0 + c4 * 4);
    bf16x4 o;
    o[0] = (bf16_t)v[0]; o[1] = (bf16_t)v[1]; o[2] = (bf16_t)v[2]; o[3] = (bf16_t)v[3];
    *(bf16x4*)(As + (size_t)r * APAD + c4 * 4) = o;
  }
}
// A tile: bf16 [M][K] -> LDS (16B loads, 16B stores)
__device__ __forceinline__ void fill_A_bf16(bf16_t* As, const bf16_t* A, int m0, int k0, int lda) {
#pragma unroll
  for (int j = 0; j < 4; ++j) {
    int idx = threadIdx.x + j * THREADS;   // 1024 bf16x8 chunks
    int r = idx >> 3, g = idx & 7;
    bf16x8 v = *(const bf16x8*)(A + (size_t)(m0 + r) * lda + k0 + g * 8);
    *(bf16x8*)(As + (size_t)r * APAD + g * 8) = v;
  }
}
// B tile: f32 activation [K][N] + per-channel affine -> LDS [n][K] bf16.
__device__ __forceinline__ void fill_B_aff(bf16_t* Bs, const float* B, const float* S,
                                           const float* T, int b, int k0, int n0, int K) {
  const int n  = threadIdx.x & 63;
  const int kg = threadIdx.x >> 6;   // 0..3
#pragma unroll
  for (int half = 0; half < 2; ++half) {
    int kb = half * 32 + kg * 8;
    const float* base = B + ((size_t)b * K + k0 + kb) * HW + n0 + n;
    bf16x8 o;
#pragma unroll
    for (int kk = 0; kk < 8; ++kk) {
      int c = k0 + kb + kk;
      float v = base[(size_t)kk * HW];
      o[kk] = (bf16_t)(v * S[b * K + c] + T[b * K + c]);
    }
    *(bf16x8*)(Bs + (size_t)n * BPAD + kb) = o;
  }
}
// B tile: bf16 activation [K][N] -> LDS [n][K]
__device__ __forceinline__ void fill_B_bf16(bf16_t* Bs, const bf16_t* B, int b, int k0,
                                            int n0, int K, int N) {
  const int n  = threadIdx.x & 63;
  const int kg = threadIdx.x >> 6;
#pragma unroll
  for (int half = 0; half < 2; ++half) {
    int kb = half * 32 + kg * 8;
    const bf16_t* base = B + ((size_t)b * K + k0 + kb) * (size_t)N + n0 + n;
    bf16x8 o;
#pragma unroll
    for (int kk = 0; kk < 8; ++kk) o[kk] = base[(size_t)kk * N];
    *(bf16x8*)(Bs + (size_t)n * BPAD + kb) = o;
  }
}

// ---------------------------------------------------------------------------
// GEMM: out[m,n] = sum_c W[m,c] * (act[c,n]*S + T) + bias[m]   (q and k/v)
// split selection hoisted to block scope (MB=128 divides split=256).
// ---------------------------------------------------------------------------
__global__ void gemm_qkv_kernel(const float* __restrict__ W, const float* bias,
                                const float* __restrict__ act, const float* S, const float* T,
                                bf16_t* out1, bf16_t* out2, int split, int K) {
  __shared__ bf16_t As[MB * APAD];
  __shared__ bf16_t Bs[NB * BPAD];
  const int b = blockIdx.z, m0 = blockIdx.y * MB, n0 = blockIdx.x * NB;
  v8f acc[2][2]; zero_acc(acc);
  for (int k0 = 0; k0 < K; k0 += KB) {
    if (k0 + KB < K)
      __builtin_prefetch(act + ((size_t)b * K + k0 + KB) * HW + n0 + (threadIdx.x & 63), 0, 1);
    fill_A_f32(As, W, m0, k0, K);
    fill_B_aff(Bs, act, S, T, b, k0, n0, K);
    __syncthreads();
    wmma_block_compute(As, Bs, acc);
    __syncthreads();
  }
  bf16_t* dst = (m0 < split) ? out1 : out2;
  const int mb = (m0 < split) ? m0 : m0 - split;
  const int lane = threadIdx.x & 31, wid = threadIdx.x >> 5;
  const int wm = wid >> 1, wn = wid & 1;
  const int colL = lane & 15, rb = (lane >> 4) * 8;
#pragma unroll
  for (int im = 0; im < 2; ++im)
#pragma unroll
    for (int in = 0; in < 2; ++in)
#pragma unroll
      for (int i = 0; i < 8; ++i) {
        int lr = wm * 32 + im * 16 + rb + i;
        int gc = n0 + wn * 32 + in * 16 + colL;
        float v = acc[im][in][i] + bias[m0 + lr];
        dst[((size_t)b * 256 + mb + lr) * HW + gc] = (bf16_t)v;
      }
}

// ---------------------------------------------------------------------------
// Anchor conv as im2col GEMM; epilogue stores TRANSPOSED: afT[b][anchor][o].
// ---------------------------------------------------------------------------
__global__ void gemm_anchor_kernel(const float* __restrict__ aw, const float* ab,
                                   const float* __restrict__ x, const float* __restrict__ p,
                                   const float* st, const float* tt, bf16_t* afT) {
  __shared__ bf16_t As[MB * APAD];
  __shared__ bf16_t Bs[NB * BPAD];
  const int b = blockIdx.z, m0 = blockIdx.y * MB, n0 = blockIdx.x * NB;
  const int nn = threadIdx.x & 63;
  const int kg = threadIdx.x >> 6;
  const int n  = n0 + nn;
  const int ay = n >> 4, ax = n & 15;
  v8f acc[2][2]; zero_acc(acc);
  for (int k0 = 0; k0 < KIM; k0 += KB) {
    fill_A_f32(As, aw, m0, k0, KIM);
#pragma unroll
    for (int half = 0; half < 2; ++half) {
      int kb = half * 32 + kg * 8;
      bf16x8 o;
#pragma unroll
      for (int kk = 0; kk < 8; ++kk) {
        int kgl = k0 + kb + kk;
        int c = kgl >> 4, s2 = kgl & 15;
        int sy = s2 >> 2, sxx = s2 & 3;
        const float* src = (c < 256) ? x : p;
        int ch = c & 255;
        float v = src[((size_t)b * CC1 + ch) * HW + (ay * 4 + sy) * 64 + ax * 4 + sxx];
        o[kk] = (bf16_t)(v * st[b * 512 + c] + tt[b * 512 + c]);
      }
      *(bf16x8*)(Bs + (size_t)nn * BPAD + kb) = o;
    }
    __syncthreads();
    wmma_block_compute(As, Bs, acc);
    __syncthreads();
  }
  const int lane = threadIdx.x & 31, wid = threadIdx.x >> 5;
  const int wm = wid >> 1, wn = wid & 1;
  const int colL = lane & 15, rb = (lane >> 4) * 8;
#pragma unroll
  for (int im = 0; im < 2; ++im)
#pragma unroll
    for (int in = 0; in < 2; ++in)
#pragma unroll
      for (int i = 0; i < 8; ++i) {
        int gr = m0 + wm * 32 + im * 16 + rb + i;   // out channel
        int gc = n0 + wn * 32 + in * 16 + colL;     // anchor
        afT[((size_t)b * NANCH + gc) * CC1 + gr] = (bf16_t)(acc[im][in][i] + ab[gr]);
      }
}

// ---------------------------------------------------------------------------
// Logits: L[a,n] = ratio * sum_c afT[a,c] * act[c,n]   (act = q or k, bf16)
// ---------------------------------------------------------------------------
__global__ void gemm_logits_kernel(const bf16_t* __restrict__ afT,
                                   const bf16_t* __restrict__ act, float* __restrict__ L) {
  __shared__ bf16_t As[MB * APAD];
  __shared__ bf16_t Bs[NB * BPAD];
  const int b = blockIdx.z, m0 = blockIdx.y * MB, n0 = blockIdx.x * NB;
  v8f acc[2][2]; zero_acc(acc);
  for (int k0 = 0; k0 < CC1; k0 += KB) {
    if (k0 + KB < CC1)
      __builtin_prefetch(act + ((size_t)b * CC1 + k0 + KB) * HW + n0 + (threadIdx.x & 63), 0, 1);
    fill_A_bf16(As, afT + (size_t)b * NANCH * CC1, m0, k0, CC1);
    fill_B_bf16(Bs, act, b, k0, n0, CC1, HW);
    __syncthreads();
    wmma_block_compute(As, Bs, acc);
    __syncthreads();
  }
  const int lane = threadIdx.x & 31, wid = threadIdx.x >> 5;
  const int wm = wid >> 1, wn = wid & 1;
  const int colL = lane & 15, rb = (lane >> 4) * 8;
#pragma unroll
  for (int im = 0; im < 2; ++im)
#pragma unroll
    for (int in = 0; in < 2; ++in)
#pragma unroll
      for (int i = 0; i < 8; ++i) {
        int gr = m0 + wm * 32 + im * 16 + rb + i;
        int gc = n0 + wn * 32 + in * 16 + colL;
        L[((size_t)b * NANCH + gr) * HW + gc] = RATIO * acc[im][in][i];
      }
}

// softmax over the anchor axis (columns of L[a][n]); coalesced across threads
__global__ void softmax_col_kernel(const float* __restrict__ L, bf16_t* __restrict__ w1T) {
  const int b = blockIdx.y;
  const int n = blockIdx.x * THREADS + threadIdx.x;
  const float* base = L + (size_t)b * NANCH * HW + n;
  float mx = -3.4e38f;
  for (int a = 0; a < NANCH; ++a) mx = fmaxf(mx, base[(size_t)a * HW]);
  float sum = 0.f;
  for (int a = 0; a < NANCH; ++a) sum += __expf(base[(size_t)a * HW] - mx);
  float inv = 1.0f / sum;
  bf16_t* ob = w1T + (size_t)b * NANCH * HW + n;
  for (int a = 0; a < NANCH; ++a)
    ob[(size_t)a * HW] = (bf16_t)(__expf(base[(size_t)a * HW] - mx) * inv);
}

// softmax over the pixel axis (rows of L[a][n])
__global__ void softmax_row_kernel(const float* __restrict__ L, bf16_t* __restrict__ w2) {
  const int b = blockIdx.y, a = blockIdx.x;
  const float* row = L + ((size_t)b * NANCH + a) * HW;
  float v[16];
  float mx = -3.4e38f;
#pragma unroll
  for (int j = 0; j < 16; ++j) { v[j] = row[threadIdx.x + j * THREADS]; mx = fmaxf(mx, v[j]); }
  __shared__ float sh[THREADS];
  sh[threadIdx.x] = mx; __syncthreads();
  for (int off = THREADS / 2; off > 0; off >>= 1) {
    if ((int)threadIdx.x < off) sh[threadIdx.x] = fmaxf(sh[threadIdx.x], sh[threadIdx.x + off]);
    __syncthreads();
  }
  mx = sh[0]; __syncthreads();
  float s = 0.f;
#pragma unroll
  for (int j = 0; j < 16; ++j) { v[j] = __expf(v[j] - mx); s += v[j]; }
  sh[threadIdx.x] = s; __syncthreads();
  for (int off = THREADS / 2; off > 0; off >>= 1) {
    if ((int)threadIdx.x < off) sh[threadIdx.x] += sh[threadIdx.x + off];
    __syncthreads();
  }
  float inv = 1.0f / sh[0];
  bf16_t* orow = w2 + ((size_t)b * NANCH + a) * HW;
#pragma unroll
  for (int j = 0; j < 16; ++j) orow[threadIdx.x + j * THREADS] = (bf16_t)(v[j] * inv);
}

// ---------------------------------------------------------------------------
// h[c,a] = sum_n v[c,n] * w2[a,n]   (w2 row direction == K: pure 16B copies)
// ---------------------------------------------------------------------------
__global__ void gemm_h_kernel(const bf16_t* __restrict__ vbuf, const bf16_t* __restrict__ w2,
                              bf16_t* __restrict__ h) {
  __shared__ bf16_t As[MB * APAD];
  __shared__ bf16_t Bs[NB * BPAD];
  const int b = blockIdx.z, m0 = blockIdx.y * MB, n0 = blockIdx.x * NB;
  const int col = threadIdx.x & 63;
  const int kg  = threadIdx.x >> 6;
  v8f acc[2][2]; zero_acc(acc);
  for (int k0 = 0; k0 < HW; k0 += KB) {
    if (k0 + KB < HW)
      __builtin_prefetch(vbuf + ((size_t)b * CC1 + m0 + (threadIdx.x >> 1)) * HW + k0 + KB, 0, 1);
#pragma unroll
    for (int half = 0; half < 2; ++half) {
      int kb = half * 32 + kg * 8;
      bf16x8 o = *(const bf16x8*)(w2 + ((size_t)b * NANCH + n0 + col) * HW + k0 + kb);
      *(bf16x8*)(Bs + (size_t)col * BPAD + kb) = o;
    }
    fill_A_bf16(As, vbuf + (size_t)b * CC1 * HW, m0, k0, HW);
    __syncthreads();
    wmma_block_compute(As, Bs, acc);
    __syncthreads();
  }
  const int lane = threadIdx.x & 31, wid = threadIdx.x >> 5;
  const int wm = wid >> 1, wn = wid & 1;
  const int colL = lane & 15, rb = (lane >> 4) * 8;
#pragma unroll
  for (int im = 0; im < 2; ++im)
#pragma unroll
    for (int in = 0; in < 2; ++in)
#pragma unroll
      for (int i = 0; i < 8; ++i) {
        int gr = m0 + wm * 32 + im * 16 + rb + i;
        int gc = n0 + wn * 32 + in * 16 + colL;
        h[((size_t)b * CC1 + gr) * NANCH + gc] = (bf16_t)acc[im][in][i];
      }
}

// h2[c,n] = sum_a h[c,a] * w1T[a,n]
__global__ void gemm_h2_kernel(const bf16_t* __restrict__ h, const bf16_t* __restrict__ w1T,
                               bf16_t* __restrict__ h2) {
  __shared__ bf16_t As[MB * APAD];
  __shared__ bf16_t Bs[NB * BPAD];
  const int b = blockIdx.z, m0 = blockIdx.y * MB, n0 = blockIdx.x * NB;
  v8f acc[2][2]; zero_acc(acc);
  for (int k0 = 0; k0 < NANCH; k0 += KB) {
    if (k0 + KB < NANCH)
      __builtin_prefetch(w1T + ((size_t)b * NANCH + k0 + KB) * HW + n0 + (threadIdx.x & 63), 0, 1);
    fill_A_bf16(As, h + (size_t)b * CC1 * NANCH, m0, k0, NANCH);
    fill_B_bf16(Bs, w1T, b, k0, n0, NANCH, HW);
    __syncthreads();
    wmma_block_compute(As, Bs, acc);
    __syncthreads();
  }
  const int lane = threadIdx.x & 31, wid = threadIdx.x >> 5;
  const int wm = wid >> 1, wn = wid & 1;
  const int colL = lane & 15, rb = (lane >> 4) * 8;
#pragma unroll
  for (int im = 0; im < 2; ++im)
#pragma unroll
    for (int in = 0; in < 2; ++in)
#pragma unroll
      for (int i = 0; i < 8; ++i) {
        int gr = m0 + wm * 32 + im * 16 + rb + i;
        int gc = n0 + wn * 32 + in * 16 + colL;
        h2[((size_t)b * CC1 + gr) * HW + gc] = (bf16_t)acc[im][in][i];
      }
}

// out[o,n] = x[o,n] + proj_b[o] + sum_c proj_w[o,c] * h2[c,n]
__global__ void gemm_proj_kernel(const float* __restrict__ pw, const float* pb,
                                 const bf16_t* __restrict__ h2, const float* __restrict__ x,
                                 float* __restrict__ out) {
  __shared__ bf16_t As[MB * APAD];
  __shared__ bf16_t Bs[NB * BPAD];
  const int b = blockIdx.z, m0 = blockIdx.y * MB, n0 = blockIdx.x * NB;
  v8f acc[2][2]; zero_acc(acc);
  for (int k0 = 0; k0 < CC1; k0 += KB) {
    if (k0 + KB < CC1)
      __builtin_prefetch(h2 + ((size_t)b * CC1 + k0 + KB) * HW + n0 + (threadIdx.x & 63), 0, 1);
    fill_A_f32(As, pw, m0, k0, CC1);
    fill_B_bf16(Bs, h2, b, k0, n0, CC1, HW);
    __syncthreads();
    wmma_block_compute(As, Bs, acc);
    __syncthreads();
  }
  const int lane = threadIdx.x & 31, wid = threadIdx.x >> 5;
  const int wm = wid >> 1, wn = wid & 1;
  const int colL = lane & 15, rb = (lane >> 4) * 8;
#pragma unroll
  for (int im = 0; im < 2; ++im)
#pragma unroll
    for (int in = 0; in < 2; ++in)
#pragma unroll
      for (int i = 0; i < 8; ++i) {
        int gr = m0 + wm * 32 + im * 16 + rb + i;
        int gc = n0 + wn * 32 + in * 16 + colL;
        size_t idx = ((size_t)b * CC1 + gr) * HW + gc;
        out[idx] = acc[im][in][i] + pb[gr] + x[idx];
      }
}

// ---------------------------------------------------------------------------
extern "C" void kernel_launch(void* const* d_in, const int* in_sizes, int n_in,
                              void* d_out, int out_size, void* d_ws, size_t ws_size,
                              hipStream_t stream) {
  (void)in_sizes; (void)n_in; (void)out_size; (void)ws_size;
  const float* x      = (const float*)d_in[0];
  const float* p      = (const float*)d_in[1];
  const float* nq_g   = (const float*)d_in[2];
  const float* nq_b   = (const float*)d_in[3];
  const float* q_w    = (const float*)d_in[4];
  const float* q_b    = (const float*)d_in[5];
  const float* nkv_g  = (const float*)d_in[6];
  const float* nkv_b  = (const float*)d_in[7];
  const float* kv_w   = (const float*)d_in[8];
  const float* kv_b   = (const float*)d_in[9];
  const float* na_g   = (const float*)d_in[10];
  const float* na_b   = (const float*)d_in[11];
  const float* a_w    = (const float*)d_in[12];
  const float* a_b    = (const float*)d_in[13];
  const float* proj_w = (const float*)d_in[14];
  const float* proj_b = (const float*)d_in[15];
  float* out = (float*)d_out;

  char* w = (char*)d_ws;
  auto take = [&](size_t bytes) -> char* {
    char* r = w; w += (bytes + 255) & ~(size_t)255; return r;
  };
  float* sx = (float*)take((size_t)BATCH * CC1 * 4);
  float* tx = (float*)take((size_t)BATCH * CC1 * 4);
  float* sp = (float*)take((size_t)BATCH * CC1 * 4);
  float* tp = (float*)take((size_t)BATCH * CC1 * 4);
  float* st = (float*)take((size_t)BATCH * 512 * 4);
  float* tt = (float*)take((size_t)BATCH * 512 * 4);
  float*  Lbuf = (float*)take((size_t)BATCH * NANCH * HW * 4);   // 64 MB, reused
  bf16_t* qbuf = (bf16_t*)take((size_t)BATCH * CC1 * HW * 2);    // 32 MB
  bf16_t* kbuf = (bf16_t*)take((size_t)BATCH * CC1 * HW * 2);
  bf16_t* vbuf = (bf16_t*)take((size_t)BATCH * CC1 * HW * 2);
  bf16_t* w1T  = (bf16_t*)take((size_t)BATCH * NANCH * HW * 2);
  bf16_t* w2   = (bf16_t*)take((size_t)BATCH * NANCH * HW * 2);
  bf16_t* h2b  = (bf16_t*)take((size_t)BATCH * CC1 * HW * 2);
  bf16_t* afT  = (bf16_t*)take((size_t)BATCH * NANCH * CC1 * 2); // 2 MB
  bf16_t* hbuf = (bf16_t*)take((size_t)BATCH * CC1 * NANCH * 2); // 2 MB

  // 1. group-norm stats for all three norms
  gn_stats_kernel<<<dim3(32, BATCH, 3), THREADS, 0, stream>>>(
      x, p, nq_g, nq_b, nkv_g, nkv_b, na_g, na_b, sx, tx, sp, tp, st, tt);

  // 2. q = q_w * gn(x) + q_b
  gemm_qkv_kernel<<<dim3(HW / NB, CC1 / MB, BATCH), THREADS, 0, stream>>>(
      q_w, q_b, x, sx, tx, qbuf, nullptr, 256, CC1);

  // 3. kv = kv_w * gn(p) + kv_b  (rows 0..255 -> k, 256..511 -> v)
  gemm_qkv_kernel<<<dim3(HW / NB, 512 / MB, BATCH), THREADS, 0, stream>>>(
      kv_w, kv_b, p, sp, tp, kbuf, vbuf, 256, CC1);

  // 4. anchors: stride-4 4x4 conv over gn(concat(x,p)) -> afT[b][a][c]
  gemm_anchor_kernel<<<dim3(NANCH / NB, CC1 / MB, BATCH), THREADS, 0, stream>>>(
      a_w, a_b, x, p, st, tt, afT);

  // 5. L1[a][n] = ratio * afT x q ; softmax over a (columns) -> w1T
  gemm_logits_kernel<<<dim3(HW / NB, NANCH / MB, BATCH), THREADS, 0, stream>>>(afT, qbuf, Lbuf);
  softmax_col_kernel<<<dim3(HW / THREADS, BATCH), THREADS, 0, stream>>>(Lbuf, w1T);

  // 6. L2[a][n] = ratio * afT x k ; softmax over n (rows) -> w2
  gemm_logits_kernel<<<dim3(HW / NB, NANCH / MB, BATCH), THREADS, 0, stream>>>(afT, kbuf, Lbuf);
  softmax_row_kernel<<<dim3(NANCH, BATCH), THREADS, 0, stream>>>(Lbuf, w2);

  // 7. h[c][a] = v x w2^T
  gemm_h_kernel<<<dim3(NANCH / NB, CC1 / MB, BATCH), THREADS, 0, stream>>>(vbuf, w2, hbuf);

  // 8. h2[c][n] = h x w1T
  gemm_h2_kernel<<<dim3(HW / NB, CC1 / MB, BATCH), THREADS, 0, stream>>>(hbuf, w1T, h2b);

  // 9. out = x + proj_w x h2 + proj_b
  gemm_proj_kernel<<<dim3(HW / NB, CC1 / MB, BATCH), THREADS, 0, stream>>>(
      proj_w, proj_b, h2b, x, out);
}
